// DLSTM_7816840479034
// MI455X (gfx1250) — matrix operand order repaired and checked by reading, not compiled, not run on hardware
//
#include <hip/hip_runtime.h>
#include <hip/hip_bf16.h>

#define BATCH 128
#define SEQ   256
#define ISZ   512
#define HSZ   1024

// LDS staging geometry: per K-stage we stage a [4 gates][16 rows][32 k] bf16 tile.
// Rows padded 64B->80B so ds_load_b128 fragment reads hit distinct banks.
#define ROWB   80
#define GATEB  (16 * ROWB)        // 1280 B per gate tile
#define BUFB   (4 * GATEB)        // 5120 B per stage buffer (x2 buffers)

typedef __bf16 bf16_t;
typedef __attribute__((ext_vector_type(8)))  __bf16 v8bf;
typedef __attribute__((ext_vector_type(16))) __bf16 v16bf;
typedef __attribute__((ext_vector_type(8)))  float  v8f;

// ---------- helpers ----------

__device__ __forceinline__ unsigned short f32_to_bf16_bits(float f) {
    unsigned u = __float_as_uint(f);
    u += 0x7FFFu + ((u >> 16) & 1u);   // round-to-nearest-even
    return (unsigned short)(u >> 16);
}

__device__ __forceinline__ float sigmoidf_(float x) {
    return 1.0f / (1.0f + __expf(-x));
}
__device__ __forceinline__ float tanhf_(float x) {
    return 2.0f / (1.0f + __expf(-2.0f * x)) - 1.0f;
}

// LDS-only workgroup barrier: orders DS ops across the block WITHOUT draining
// outstanding global loads (unlike __syncthreads(), which waits loadcnt->0 and
// would truncate the global prefetch pipeline).  Per CDNA5 ISA:
// split barrier = signal(-1) + wait(-1); DS visibility needs s_wait_dscnt 0.
__device__ __forceinline__ void block_sync_lds() {
    asm volatile(
        "s_wait_dscnt 0x0\n\t"
        "s_barrier_signal -1\n\t"
        "s_barrier_wait -1"
        ::: "memory");
}

// A fragment, 16x32 bf16 (M x K): lane half h holds K chunks [8h..8h+7] and [8h+16..8h+23]
__device__ __forceinline__ v16bf load_a_frag(const bf16_t* __restrict__ row_ptr, int k0) {
    v8bf lo = *(const v8bf*)(row_ptr + k0);
    v8bf hi = *(const v8bf*)(row_ptr + k0 + 16);
    return __builtin_shufflevector(lo, hi, 0,1,2,3,4,5,6,7,8,9,10,11,12,13,14,15);
}

// B fragment from LDS tile: lane half h reads 32 contiguous bytes at row (lane&15)
__device__ __forceinline__ v16bf load_b_lds(const char* p) {
    v8bf lo = *(const v8bf*)(p);
    v8bf hi = *(const v8bf*)(p + 16);
    return __builtin_shufflevector(lo, hi, 0,1,2,3,4,5,6,7,8,9,10,11,12,13,14,15);
}

// GEMM phase with LDS-staged, block-shared B tiles (double buffered) and
// depth-2 register prefetch of the per-wave A fragments.
//   arow : per-lane A row base (global bf16)
//   bptr : per-thread staging pointer into W (global bf16), advances by k
//   lds  : 2*BUFB byte staging area
//   st_off: per-thread LDS store offset inside a buffer
//   ld_off: per-lane LDS load offset inside a buffer (gate offset added here)
template<int K>
__device__ __forceinline__ void gemm_phase_lds(v8f* acc,
                                               const bf16_t* __restrict__ arow,
                                               const bf16_t* __restrict__ bptr,
                                               char* lds, int st_off, int ld_off,
                                               int ka0) {
    constexpr int NB = K / 32;
    v8bf  breg[2];
    v16bf a[3];

    // prologue: stage 0 into buf0, prefetch stage 1 regs, prefetch A 0/1
    breg[0] = *(const v8bf*)(bptr + 0);
    *(v8bf*)(lds + 0 * BUFB + st_off) = breg[0];
    if (NB > 1) breg[1] = *(const v8bf*)(bptr + 32);
    a[0] = load_a_frag(arow, 0 + ka0);
    if (NB > 1) a[1] = load_a_frag(arow, 32 + ka0);
    block_sync_lds();

    #pragma unroll
    for (int i = 0; i < NB; ++i) {
        if (i + 1 < NB) {                        // store stage i+1 (held in breg[(i+1)&1])
            *(v8bf*)(lds + ((i + 1) & 1) * BUFB + st_off) = breg[(i + 1) & 1];
        }
        if (i + 2 < NB) {                        // prefetch stage i+2 from global
            breg[i & 1] = *(const v8bf*)(bptr + (i + 2) * 32);
            a[(i + 2) % 3] = load_a_frag(arow, (i + 2) * 32 + ka0);
        }
        const char* bb = lds + (i & 1) * BUFB + ld_off;
        v16bf b0 = load_b_lds(bb + 0 * GATEB);
        v16bf b1 = load_b_lds(bb + 1 * GATEB);
        v16bf b2 = load_b_lds(bb + 2 * GATEB);
        v16bf b3 = load_b_lds(bb + 3 * GATEB);
        const int c = i % 3;
        acc[0] = __builtin_amdgcn_wmma_f32_16x16x32_bf16(false, a[c], false, b0, (short)0, acc[0], false, false);
        acc[1] = __builtin_amdgcn_wmma_f32_16x16x32_bf16(false, a[c], false, b1, (short)0, acc[1], false, false);
        acc[2] = __builtin_amdgcn_wmma_f32_16x16x32_bf16(false, a[c], false, b2, (short)0, acc[2], false, false);
        acc[3] = __builtin_amdgcn_wmma_f32_16x16x32_bf16(false, a[c], false, b3, (short)0, acc[3], false, false);
        block_sync_lds();
    }
}

// ---------- utility kernels ----------

__global__ void f32_to_bf16_kernel(const float* __restrict__ in,
                                   unsigned short* __restrict__ out, size_t n) {
    size_t i = (size_t)blockIdx.x * blockDim.x + threadIdx.x;
    size_t stride = (size_t)gridDim.x * blockDim.x;
    for (; i < n; i += stride) out[i] = f32_to_bf16_bits(in[i]);
}

__global__ void zero_u32_kernel(unsigned* __restrict__ p, size_t n_words) {
    size_t i = (size_t)blockIdx.x * blockDim.x + threadIdx.x;
    size_t stride = (size_t)gridDim.x * blockDim.x;
    for (; i < n_words; i += stride) p[i] = 0u;
}

// ---------- LSTM step kernels ----------
// Grid: 64 blocks (one per 16-wide N tile) x 256 threads (8 waves = 8 M tiles).
// All waves in a block share the same weight rows -> B staged once in LDS.

__global__ __launch_bounds__(256, 2) void lstm_step0(
    const bf16_t* __restrict__ xbf,   // [B][T][ISZ] bf16
    const bf16_t* __restrict__ wih,   // [4H][ISZ]   bf16
    const bf16_t* __restrict__ whh,   // [4H][HSZ]   bf16
    const float*  __restrict__ bih, const float* __restrict__ bhh,
    const bf16_t* __restrict__ h_in,  // [B][HSZ]
    unsigned short* __restrict__ h_out, // [B][HSZ] bf16 bits
    float*        __restrict__ c,     // [B][HSZ]
    int t)
{
    __shared__ __align__(16) char smem[2 * BUFB];

    const int local = threadIdx.x;
    const int lane  = local & 31;
    const int lr    = lane & 15;
    const int half  = lane >> 4;
    const int m_base = (local >> 5) << 4;      // wave id = M tile
    const int n_base = blockIdx.x << 4;        // block id = N tile
    const int ka0 = half * 8;

    // staging mapping: thread = (gate, row, chunk)
    const int sg  = local >> 6;
    const int sr  = (local >> 2) & 15;
    const int scq = local & 3;
    const int st_off = sg * GATEB + sr * ROWB + (((scq & 1) << 1) | (scq >> 1)) * 16;
    const int ld_off = lr * ROWB + half * 32;

    v8f acc[4];
    acc[0] = (v8f)0.0f; acc[1] = (v8f)0.0f; acc[2] = (v8f)0.0f; acc[3] = (v8f)0.0f;

    // phase 1: x_t @ Wih^T   (K = ISZ)
    {
        const bf16_t* arow = xbf + ((size_t)(m_base + lr) * SEQ + t) * ISZ;
        const bf16_t* bptr = wih + (size_t)(n_base + sr + sg * HSZ) * ISZ + scq * 8;
        gemm_phase_lds<ISZ>(acc, arow, bptr, smem, st_off, ld_off, ka0);
    }
    // phase 2: h_{t-1} @ Whh^T   (K = HSZ)
    {
        const bf16_t* arow = h_in + (size_t)(m_base + lr) * HSZ;
        const bf16_t* bptr = whh + (size_t)(n_base + sr + sg * HSZ) * HSZ + scq * 8;
        gemm_phase_lds<HSZ>(acc, arow, bptr, smem, st_off, ld_off, ka0);
    }

    // epilogue: bias + cell update
    const int n = n_base + lr;                  // C layout: N = lane & 15
    float bsum[4];
    #pragma unroll
    for (int g = 0; g < 4; ++g) bsum[g] = bih[n + g * HSZ] + bhh[n + g * HSZ];

    #pragma unroll
    for (int v = 0; v < 8; ++v) {
        const int m = m_base + v + 8 * half;    // C layout: M = v + 8*(lane>=16)
        const size_t idx = (size_t)m * HSZ + n;
        const float gi = acc[0][v] + bsum[0];
        const float gf = acc[1][v] + bsum[1];
        const float gg = acc[2][v] + bsum[2];
        const float go = acc[3][v] + bsum[3];
        const float cn = sigmoidf_(gf) * c[idx] + sigmoidf_(gi) * tanhf_(gg);
        const float h  = sigmoidf_(go) * tanhf_(cn);
        c[idx] = cn;
        h_out[idx] = f32_to_bf16_bits(h);
    }
}

__global__ __launch_bounds__(256, 2) void lstm_step1(
    const bf16_t* __restrict__ h0,    // [B][HSZ] bf16 (layer-0 output at time t)
    const bf16_t* __restrict__ wih,   // [4H][HSZ]
    const bf16_t* __restrict__ whh,   // [4H][HSZ]
    const float*  __restrict__ bih, const float* __restrict__ bhh,
    const bf16_t* __restrict__ h_in,  // [B][HSZ] (layer-1 h_{t-1})
    unsigned short* __restrict__ h_out,
    float*        __restrict__ c,     // [B][HSZ]
    float*        __restrict__ out,   // [B][T][HSZ] f32
    int t)
{
    __shared__ __align__(16) char smem[2 * BUFB];

    const int local = threadIdx.x;
    const int lane  = local & 31;
    const int lr    = lane & 15;
    const int half  = lane >> 4;
    const int m_base = (local >> 5) << 4;
    const int n_base = blockIdx.x << 4;
    const int ka0 = half * 8;

    const int sg  = local >> 6;
    const int sr  = (local >> 2) & 15;
    const int scq = local & 3;
    const int st_off = sg * GATEB + sr * ROWB + (((scq & 1) << 1) | (scq >> 1)) * 16;
    const int ld_off = lr * ROWB + half * 32;

    v8f acc[4];
    acc[0] = (v8f)0.0f; acc[1] = (v8f)0.0f; acc[2] = (v8f)0.0f; acc[3] = (v8f)0.0f;

    // phase 1: h0_t @ Wih1^T
    {
        const bf16_t* arow = h0 + (size_t)(m_base + lr) * HSZ;
        const bf16_t* bptr = wih + (size_t)(n_base + sr + sg * HSZ) * HSZ + scq * 8;
        gemm_phase_lds<HSZ>(acc, arow, bptr, smem, st_off, ld_off, ka0);
    }
    // phase 2: h1_{t-1} @ Whh1^T
    {
        const bf16_t* arow = h_in + (size_t)(m_base + lr) * HSZ;
        const bf16_t* bptr = whh + (size_t)(n_base + sr + sg * HSZ) * HSZ + scq * 8;
        gemm_phase_lds<HSZ>(acc, arow, bptr, smem, st_off, ld_off, ka0);
    }

    const int n = n_base + lr;
    float bsum[4];
    #pragma unroll
    for (int g = 0; g < 4; ++g) bsum[g] = bih[n + g * HSZ] + bhh[n + g * HSZ];

    #pragma unroll
    for (int v = 0; v < 8; ++v) {
        const int m = m_base + v + 8 * half;
        const size_t idx = (size_t)m * HSZ + n;
        const float gi = acc[0][v] + bsum[0];
        const float gf = acc[1][v] + bsum[1];
        const float gg = acc[2][v] + bsum[2];
        const float go = acc[3][v] + bsum[3];
        const float cn = sigmoidf_(gf) * c[idx] + sigmoidf_(gi) * tanhf_(gg);
        const float h  = sigmoidf_(go) * tanhf_(cn);
        c[idx] = cn;
        h_out[idx] = f32_to_bf16_bits(h);
        out[((size_t)m * SEQ + t) * HSZ + n] = h;   // y[b][t][h]
    }
}

// ---------- host launch ----------

extern "C" void kernel_launch(void* const* d_in, const int* in_sizes, int n_in,
                              void* d_out, int out_size, void* d_ws, size_t ws_size,
                              hipStream_t stream) {
    (void)in_sizes; (void)n_in; (void)out_size; (void)ws_size;
    const float* x    = (const float*)d_in[0];
    const float* Wih0 = (const float*)d_in[1];
    const float* Whh0 = (const float*)d_in[2];
    const float* bih0 = (const float*)d_in[3];
    const float* bhh0 = (const float*)d_in[4];
    const float* Wih1 = (const float*)d_in[5];
    const float* Whh1 = (const float*)d_in[6];
    const float* bih1 = (const float*)d_in[7];
    const float* bhh1 = (const float*)d_in[8];

    char* w = (char*)d_ws;
    size_t off = 0;
    float* c0 = (float*)(w + off);                off += (size_t)BATCH * HSZ * 4;
    float* c1 = (float*)(w + off);                off += (size_t)BATCH * HSZ * 4;
    unsigned short* h1buf = (unsigned short*)(w + off); off += (size_t)2 * BATCH * HSZ * 2;
    unsigned short* h0seq = (unsigned short*)(w + off); off += (size_t)(SEQ + 1) * BATCH * HSZ * 2;
    unsigned short* xbf   = (unsigned short*)(w + off); off += (size_t)BATCH * SEQ * ISZ * 2;
    unsigned short* wih0b = (unsigned short*)(w + off); off += (size_t)4 * HSZ * ISZ * 2;
    unsigned short* whh0b = (unsigned short*)(w + off); off += (size_t)4 * HSZ * HSZ * 2;
    unsigned short* wih1b = (unsigned short*)(w + off); off += (size_t)4 * HSZ * HSZ * 2;
    unsigned short* whh1b = (unsigned short*)(w + off); off += (size_t)4 * HSZ * HSZ * 2;

    // 1) convert inputs/weights to bf16 (row-major, fragment-friendly)
    f32_to_bf16_kernel<<<2048, 256, 0, stream>>>(x,    xbf,   (size_t)BATCH * SEQ * ISZ);
    f32_to_bf16_kernel<<<1024, 256, 0, stream>>>(Wih0, wih0b, (size_t)4 * HSZ * ISZ);
    f32_to_bf16_kernel<<<1024, 256, 0, stream>>>(Whh0, whh0b, (size_t)4 * HSZ * HSZ);
    f32_to_bf16_kernel<<<1024, 256, 0, stream>>>(Wih1, wih1b, (size_t)4 * HSZ * HSZ);
    f32_to_bf16_kernel<<<1024, 256, 0, stream>>>(Whh1, whh1b, (size_t)4 * HSZ * HSZ);

    // 2) zero initial state: c0+c1 (contiguous f32), h1 slot 0, h0seq slot 0
    zero_u32_kernel<<<256, 256, 0, stream>>>((unsigned*)c0,    (size_t)BATCH * HSZ * 2);
    zero_u32_kernel<<<64,  256, 0, stream>>>((unsigned*)h1buf, (size_t)BATCH * HSZ / 2);
    zero_u32_kernel<<<64,  256, 0, stream>>>((unsigned*)h0seq, (size_t)BATCH * HSZ / 2);

    // 3) recurrence: one kernel per (timestep, layer) for device-wide ordering
    for (int t = 0; t < SEQ; ++t) {
        const bf16_t* h0_in  = (const bf16_t*)(h0seq + (size_t)t * BATCH * HSZ);
        unsigned short* h0_out = h0seq + (size_t)(t + 1) * BATCH * HSZ;
        lstm_step0<<<64, 256, 0, stream>>>((const bf16_t*)xbf, (const bf16_t*)wih0b,
                                           (const bf16_t*)whh0b, bih0, bhh0,
                                           h0_in, h0_out, c0, t);

        const bf16_t* h1_in  = (const bf16_t*)(h1buf + (size_t)(t & 1) * BATCH * HSZ);
        unsigned short* h1_out = h1buf + (size_t)((t + 1) & 1) * BATCH * HSZ;
        lstm_step1<<<64, 256, 0, stream>>>((const bf16_t*)h0_out, (const bf16_t*)wih1b,
                                           (const bf16_t*)whh1b, bih1, bhh1,
                                           h1_in, h1_out, c1, (float*)d_out, t);
    }
}
